// Robustkt_3487513444384
// MI455X (gfx1250) — compile-verified, matrix operand order checked by reading, not run on hardware
//
#include <hip/hip_runtime.h>
#include <hip/hip_bf16.h>
#include <math.h>

// ---------------------------------------------------------------------------
// Shapes (fixed by the reference)
// ---------------------------------------------------------------------------
#define BSZ 8
#define SEQT 1024
#define DIM 512
#define NH 8
#define DKH 64
#define FFD 2048
#define KSZ 5
#define NL 6
#define ROWS (BSZ * SEQT) // 8192

typedef __attribute__((ext_vector_type(16))) __bf16 v16bf;
typedef __attribute__((ext_vector_type(8)))  float  v8f;
typedef __attribute__((ext_vector_type(4)))  int    v4i;

// Optional CDNA5 async global->LDS path (ASYNCcnt-tracked). Guarded so the
// kernel still compiles (with the manual staging fallback) if this toolchain
// doesn't declare the builtins. Prototype (from hipcc diagnostics):
//   void __builtin_amdgcn_global_load_async_to_lds_b128(
//       v4i addrspace(1)* src, v4i addrspace(3)* dst, imm int off, imm int cpol)
#if defined(__has_builtin)
#if __has_builtin(__builtin_amdgcn_global_load_async_to_lds_b128) && \
    __has_builtin(__builtin_amdgcn_s_wait_asynccnt)
#define USE_ASYNC_LDS 1
#endif
#endif

#ifdef USE_ASYNC_LDS
typedef __attribute__((address_space(1))) v4i g_v4i; // global
typedef __attribute__((address_space(3))) v4i l_v4i; // LDS
#endif

__device__ __forceinline__ float wave_sum(float v) {
#pragma unroll
  for (int o = 16; o > 0; o >>= 1) v += __shfl_xor(v, o, 32);
  return v;
}
__device__ __forceinline__ float wave_max(float v) {
#pragma unroll
  for (int o = 16; o > 0; o >>= 1) v = fmaxf(v, __shfl_xor(v, o, 32));
  return v;
}

// ---------------------------------------------------------------------------
// Generic bf16 WMMA GEMM:  C[M,N](f32 or bf16) = A[M,K] * B  (+bias)(+relu)
//   A: row-major bf16, with optional per-row time shift (for causal conv taps)
//   B: element (k,n) at B[k*ldbk + n*ldbn]  (supports row-major & conv layout)
// Tile: 128(M) x 64(N) per 256-thread block; 8 waves, wave w owns rows w*16..+16.
// Inner step K=32 -> one v_wmma_f32_16x16x32_bf16 per 16x16 subtile.
// A staging uses GLOBAL_LOAD_ASYNC_TO_LDS_B128 when available (Trows==0 path);
// B tile is gathered/transposed manually (strided weight layouts).
// ---------------------------------------------------------------------------
__global__ __launch_bounds__(256) void gemm_bf16_wmma(
    const __bf16* __restrict__ A, const __bf16* __restrict__ B,
    const float* __restrict__ bias,
    float* __restrict__ Cf, __bf16* __restrict__ Ch,
    int M, int N, int K, long ldbk, long ldbn,
    int t_shift, int Trows, int accumulate, int relu) {
  __shared__ __bf16 As[128 * 32];  // 8 KB
  __shared__ __bf16 Bt[64 * 32];   // 4 KB, stored transposed: Bt[n][k]

  const int tid  = threadIdx.x;
  const int lane = tid & 31;
  const int wave = tid >> 5;
  const int m0 = blockIdx.y * 128;
  const int n0 = blockIdx.x * 64;

  v8f acc[4] = {};

  for (int k0 = 0; k0 < K; k0 += 32) {
    // ---- stage A tile (128x32), 32 bytes per thread ----
    {
      int rid  = tid >> 1;
      int colb = (tid & 1) * 16;
      int r = m0 + rid;
#ifdef USE_ASYNC_LDS
      if (Trows == 0) {
        // async DMA straight into LDS, no VGPR round-trip (ASYNCcnt)
        const __bf16* gsrc = A + (long)r * K + k0 + colb;
        __bf16* ldst = &As[rid * 32 + colb];
        __builtin_amdgcn_global_load_async_to_lds_b128(
            (g_v4i*)(void*)gsrc, (l_v4i*)(void*)ldst, 0, 0);
        __builtin_amdgcn_global_load_async_to_lds_b128(
            (g_v4i*)(void*)gsrc, (l_v4i*)(void*)ldst, 16, 0);
      } else
#endif
      {
        bool valid = true;
        long rs = r;
        if (Trows > 0) {
          int t  = r % Trows;
          int ts = t + t_shift;
          valid = (ts >= 0) && (ts < Trows);
          rs = (long)r + t_shift;
        }
        v16bf av = {};
        if (valid) av = *(const v16bf*)(A + rs * (long)K + k0 + colb);
        *(v16bf*)&As[rid * 32 + colb] = av;
      }
    }
    // ---- stage B tile transposed (64 n x 32 k), 8 scalars per thread ----
    {
      int n  = tid & 63;
      int kb = (tid >> 6) * 8;
#pragma unroll
      for (int j = 0; j < 8; j++) {
        int k = kb + j;
        Bt[n * 32 + k] = B[(long)(k0 + k) * ldbk + (long)(n0 + n) * ldbn];
      }
      // pull next K-step's B line toward the WGP (global_prefetch_b8)
      if (k0 + 32 < K)
        __builtin_prefetch(B + (long)(k0 + 32 + kb) * ldbk + (long)(n0 + n) * ldbn, 0, 3);
    }
#ifdef USE_ASYNC_LDS
    if (Trows == 0) __builtin_amdgcn_s_wait_asynccnt(0);
#endif
    __syncthreads();
    // ---- WMMA: A fragment lanes 0-15 K=0..15 / lanes 16-31 K=16..31 ----
    {
      const int mrow = wave * 16 + (lane & 15);
      const int kb   = (lane >> 4) * 16;
      v16bf a = *(const v16bf*)&As[mrow * 32 + kb];
#pragma unroll
      for (int j = 0; j < 4; j++) {
        v16bf bfr = *(const v16bf*)&Bt[(j * 16 + (lane & 15)) * 32 + kb];
        acc[j] = __builtin_amdgcn_wmma_f32_16x16x32_bf16(
            false, a, false, bfr, (short)0, acc[j], false, false);
      }
    }
    __syncthreads();
  }

  // ---- epilogue: VGPR i -> row i (lanes 0-15) / row 8+i (lanes 16-31) ----
  const int rowbase = m0 + wave * 16 + ((lane >> 4) * 8);
  const int colbase = n0 + (lane & 15);
#pragma unroll
  for (int j = 0; j < 4; j++) {
#pragma unroll
    for (int i = 0; i < 8; i++) {
      int r = rowbase + i;
      int c = colbase + j * 16;
      float val = acc[j][i];
      if (bias) val += bias[c];
      if (accumulate && Cf) val += Cf[(long)r * N + c];
      if (relu) val = fmaxf(val, 0.0f);
      if (Ch) Ch[(long)r * N + c] = (__bf16)val;
      else    Cf[(long)r * N + c] = val;
    }
  }
}

// ---------------------------------------------------------------------------
// Fused decay attention. One block = (batch b, head h, 16-query tile).
// Phase 1: S[16][1024] = mask(Q*K^T / 8)            (WMMA bf16)
// Phase 2: per-row softmax -> cumsum (wave scan) -> decay effect ->
//          second softmax -> P bf16 in LDS
// Phase 3: O[16][64] = P * V                         (WMMA bf16, per-wave
//          K-split + transposed V staging in private LDS, ds_add_f32 reduce)
// LDS: S 64KB + Ored 4KB + P 32KB + Vt 32KB = 132KB (dynamic; CDNA5 WGP has
// 320KB so this still allows 2 workgroups per WGP).
// ---------------------------------------------------------------------------
#define ATTN_SMEM (16 * 1024 * 4 + 16 * 64 * 4 + 16 * 1024 * 2 + 8 * 64 * 32 * 2)

__global__ __launch_bounds__(256) void attn_decay_wmma(
    const __bf16* __restrict__ qk, const __bf16* __restrict__ vv,
    const float* __restrict__ gammas, int layer, int mtype,
    __bf16* __restrict__ concat) {
  extern __shared__ char smem[];
  float*  S    = (float*)smem;                                   // 16x1024
  float*  Ored = (float*)(smem + 16 * 1024 * 4);                 // 16x64
  __bf16* P    = (__bf16*)(smem + 16 * 1024 * 4 + 16 * 64 * 4);  // 16x1024
  __bf16* Vt   = (__bf16*)(smem + 16 * 1024 * 4 + 16 * 64 * 4 + 16 * 1024 * 2);

  const int tid  = threadIdx.x;
  const int lane = tid & 31;
  const int wave = tid >> 5;
  const int qt0 = blockIdx.x * 16;
  const int h   = blockIdx.y;
  const int b   = blockIdx.z;
  const long base = ((long)b * SEQT) * DIM + h * DKH;

  for (int idx = tid; idx < 16 * 64; idx += 256) Ored[idx] = 0.0f;

  // ---------------- phase 1: scores ----------------
  {
    const int mr = lane & 15;
    const int kb = (lane >> 4) * 16;
    v16bf a0 = *(const v16bf*)(qk + base + (long)(qt0 + mr) * DIM + kb);
    v16bf a1 = *(const v16bf*)(qk + base + (long)(qt0 + mr) * DIM + 32 + kb);
    for (int c = 0; c < 8; c++) {
      int n0l = wave * 128 + c * 16;
      // B fragment: B[k=dk][n=key] = K[key][dk] -> contiguous dk in memory
      v16bf b0 = *(const v16bf*)(qk + base + (long)(n0l + mr) * DIM + kb);
      v16bf b1 = *(const v16bf*)(qk + base + (long)(n0l + mr) * DIM + 32 + kb);
      v8f acc = {};
      acc = __builtin_amdgcn_wmma_f32_16x16x32_bf16(false, a0, false, b0, (short)0, acc, false, false);
      acc = __builtin_amdgcn_wmma_f32_16x16x32_bf16(false, a1, false, b1, (short)0, acc, false, false);
#pragma unroll
      for (int i = 0; i < 8; i++) {
        int row = i + ((lane >> 4) * 8);
        int col = n0l + mr;
        int qg  = qt0 + row;
        bool valid = mtype ? (col <= qg) : (col < qg);
        S[row * 1024 + col] = valid ? acc[i] * 0.125f : -1e30f;
      }
    }
  }
  __syncthreads();

  // ---------------- phase 2: softmax / cumsum / decay / softmax ----------------
  {
    float gamma = -log1pf(__expf(gammas[layer * NH + h])); // -softplus
    for (int rr = wave; rr < 16; rr += 8) {
      float*  Srow = S + rr * 1024;
      __bf16* Prow = P + rr * 1024;
      int qg = qt0 + rr;
      if (mtype == 0 && qg == 0) { // fully-masked row + zero_pad -> zero output
        for (int c = 0; c < 32; c++) Prow[c * 32 + lane] = (__bf16)0.0f;
        continue;
      }
      float m = -1e30f;
      for (int c = 0; c < 32; c++) m = fmaxf(m, Srow[c * 32 + lane]);
      m = wave_max(m);
      float Z = 0.0f;
      for (int c = 0; c < 32; c++) Z += __expf(Srow[c * 32 + lane] - m);
      Z = wave_sum(Z);
      float invZ = 1.0f / Z;

      // cumsum of s_ = softmax*mask; disttot == 1; rewrite scores with effect
      float carry = 0.0f;
      float m2 = -1e30f;
      for (int c = 0; c < 32; c++) {
        int col = c * 32 + lane;
        float raw = Srow[col];
        float e = __expf(raw - m) * invZ;
        float x = e;
#pragma unroll
        for (int o = 1; o < 32; o <<= 1) {
          float tt = __shfl_up(x, o, 32);
          if (lane >= o) x += tt;
        }
        float cum = carry + x;
        carry += __shfl(x, 31, 32);
        float pos  = fabsf((float)(col - qg));
        float dist = sqrtf(fmaxf((1.0f - cum) * pos, 0.0f));
        float eff  = fminf(fmaxf(__expf(dist * gamma), 1e-5f), 1e5f);
        bool valid = mtype ? (col <= qg) : (col < qg);
        float nv = valid ? raw * eff : -1e30f;
        Srow[col] = nv;
        m2 = fmaxf(m2, nv);
      }
      m2 = wave_max(m2);
      float Z2 = 0.0f;
      for (int c = 0; c < 32; c++) {
        int col = c * 32 + lane;
        float e2 = __expf(Srow[col] - m2);
        Srow[col] = e2;
        Z2 += e2;
      }
      Z2 = wave_sum(Z2);
      float inv2 = 1.0f / Z2;
      for (int c = 0; c < 32; c++) {
        int col = c * 32 + lane;
        Prow[col] = (__bf16)(Srow[col] * inv2);
      }
    }
  }
  __syncthreads();

  // ---------------- phase 3: O = P * V (per-wave K split) ----------------
  {
    __bf16* vtw = Vt + wave * (64 * 32); // private transposed staging
    v8f acc[4] = {};
    for (int ch = 0; ch < 4; ch++) {
      int kbase = wave * 128 + ch * 32;
      { // stage V[kbase..+32][0..64) transposed: vtw[n*32 + k]
        const __bf16* vr = vv + base + (long)(kbase + lane) * DIM;
        if (ch < 3) __builtin_prefetch(vr + 32 * DIM, 0, 3); // next chunk
        v16bf t0 = *(const v16bf*)(vr + 0);
        v16bf t1 = *(const v16bf*)(vr + 16);
        v16bf t2 = *(const v16bf*)(vr + 32);
        v16bf t3 = *(const v16bf*)(vr + 48);
#pragma unroll
        for (int n = 0; n < 16; n++) vtw[(n)*32 + lane]        = t0[n];
#pragma unroll
        for (int n = 0; n < 16; n++) vtw[(n + 16) * 32 + lane] = t1[n];
#pragma unroll
        for (int n = 0; n < 16; n++) vtw[(n + 32) * 32 + lane] = t2[n];
#pragma unroll
        for (int n = 0; n < 16; n++) vtw[(n + 48) * 32 + lane] = t3[n];
      }
      const int mr = lane & 15;
      const int kb = (lane >> 4) * 16;
      v16bf a = *(const v16bf*)&P[mr * 1024 + kbase + kb];
#pragma unroll
      for (int j = 0; j < 4; j++) {
        v16bf bfr = *(const v16bf*)&vtw[(j * 16 + mr) * 32 + kb];
        acc[j] = __builtin_amdgcn_wmma_f32_16x16x32_bf16(
            false, a, false, bfr, (short)0, acc[j], false, false);
      }
    }
#pragma unroll
    for (int j = 0; j < 4; j++)
#pragma unroll
      for (int i = 0; i < 8; i++) {
        int row = i + ((lane >> 4) * 8);
        int col = j * 16 + (lane & 15);
        atomicAdd(&Ored[row * 64 + col], acc[j][i]); // ds_add_f32
      }
  }
  __syncthreads();
  for (int idx = tid; idx < 16 * 64; idx += 256) {
    int row = idx >> 6, col = idx & 63;
    concat[((long)b * SEQT + qt0 + row) * DIM + h * DKH + col] = (__bf16)Ored[idx];
  }
}

// ---------------------------------------------------------------------------
// out = LayerNorm(x + y) * g + b   (D=512, one wave per row, 8 rows/block)
// Writes f32 + bf16 copies.
// ---------------------------------------------------------------------------
__global__ __launch_bounds__(256) void add_ln_512(
    const float* __restrict__ x, const float* __restrict__ y,
    const float* __restrict__ g, const float* __restrict__ bb,
    float* __restrict__ outf, __bf16* __restrict__ outh, float eps) {
  const int lane = threadIdx.x & 31;
  const long row = (long)blockIdx.x * 8 + (threadIdx.x >> 5);
  const float* xr = x + row * 512;
  const float* yr = y + row * 512;
  float vals[16], s = 0.0f, s2 = 0.0f;
#pragma unroll
  for (int i = 0; i < 16; i++) {
    int c = lane + i * 32;
    float vvv = xr[c] + yr[c];
    vals[i] = vvv; s += vvv; s2 += vvv * vvv;
  }
  s = wave_sum(s); s2 = wave_sum(s2);
  float mean = s * (1.0f / 512.0f);
  float var  = s2 * (1.0f / 512.0f) - mean * mean;
  float rinv = rsqrtf(var + eps);
#pragma unroll
  for (int i = 0; i < 16; i++) {
    int c = lane + i * 32;
    float o = g[c] * (vals[i] - mean) * rinv + bb[c];
    outf[row * 512 + c] = o;
    outh[row * 512 + c] = (__bf16)o;
  }
}

// out = LN(trend + sb^2*(x-trend) + x)  (smooth front-end, eps 1e-12)
__global__ __launch_bounds__(256) void smooth_ln_512(
    const float* __restrict__ x, const float* __restrict__ trend,
    const float* __restrict__ sb, const float* __restrict__ g,
    const float* __restrict__ bb, float* __restrict__ outf,
    __bf16* __restrict__ outh) {
  const int lane = threadIdx.x & 31;
  const long row = (long)blockIdx.x * 8 + (threadIdx.x >> 5);
  const float* xr = x + row * 512;
  const float* tr = trend + row * 512;
  float vals[16], s = 0.0f, s2 = 0.0f;
#pragma unroll
  for (int i = 0; i < 16; i++) {
    int c = lane + i * 32;
    float xv = xr[c], tv = tr[c], sc = sb[c];
    float seq = tv + sc * sc * (xv - tv);
    float vvv = seq + xv;
    vals[i] = vvv; s += vvv; s2 += vvv * vvv;
  }
  s = wave_sum(s); s2 = wave_sum(s2);
  float mean = s * (1.0f / 512.0f);
  float var  = s2 * (1.0f / 512.0f) - mean * mean;
  float rinv = rsqrtf(var + 1e-12f);
#pragma unroll
  for (int i = 0; i < 16; i++) {
    int c = lane + i * 32;
    float o = g[c] * (vals[i] - mean) * rinv + bb[c];
    outf[row * 512 + c] = o;
    outh[row * 512 + c] = (__bf16)o;
  }
}

__global__ void cvt_f32_bf16(const float* __restrict__ in,
                             __bf16* __restrict__ out, long n) {
  long i = (long)blockIdx.x * blockDim.x + threadIdx.x;
  long stride = (long)gridDim.x * blockDim.x;
  for (; i < n; i += stride) out[i] = (__bf16)in[i];
}

// ---------------------------------------------------------------------------
// Host orchestration
// ---------------------------------------------------------------------------
extern "C" void kernel_launch(void* const* d_in, const int* in_sizes, int n_in,
                              void* d_out, int out_size, void* d_ws, size_t ws_size,
                              hipStream_t stream) {
  (void)in_sizes; (void)n_in; (void)out_size; (void)ws_size;
  const float* q_emb   = (const float*)d_in[0];
  const float* qa_emb  = (const float*)d_in[1];
  const float* conv_w  = (const float*)d_in[3];
  const float* conv_b  = (const float*)d_in[4];
  const float* sqrt_b  = (const float*)d_in[5];
  const float* sm_g    = (const float*)d_in[6];
  const float* sm_b    = (const float*)d_in[7];
  const float* k_w     = (const float*)d_in[8];
  const float* k_b     = (const float*)d_in[9];
  const float* v_w     = (const float*)d_in[10];
  const float* v_b     = (const float*)d_in[11];
  const float* out_w   = (const float*)d_in[12];
  const float* out_b   = (const float*)d_in[13];
  const float* gammas  = (const float*)d_in[14];
  const float* ln1_g   = (const float*)d_in[15];
  const float* ln1_b   = (const float*)d_in[16];
  const float* ff1_w   = (const float*)d_in[17];
  const float* ff1_b   = (const float*)d_in[18];
  const float* ff2_w   = (const float*)d_in[19];
  const float* ff2_b   = (const float*)d_in[20];
  const float* ln2_g   = (const float*)d_in[21];
  const float* ln2_b   = (const float*)d_in[22];

  char* wsp = (char*)d_ws;
  size_t off = 0;
  auto alloc = [&](size_t bytes) -> void* {
    void* p = wsp + off;
    off = (off + bytes + 255) & ~(size_t)255;
    return p;
  };
  __bf16* Wk  = (__bf16*)alloc((size_t)NL * DIM * DIM * 2);
  __bf16* Wv  = (__bf16*)alloc((size_t)NL * DIM * DIM * 2);
  __bf16* Wo  = (__bf16*)alloc((size_t)NL * DIM * DIM * 2);
  __bf16* Wf1 = (__bf16*)alloc((size_t)NL * DIM * FFD * 2);
  __bf16* Wf2 = (__bf16*)alloc((size_t)NL * FFD * DIM * 2);
  __bf16* Wc  = (__bf16*)alloc((size_t)DIM * DIM * KSZ * 2);
  __bf16* b_x  = (__bf16*)alloc((size_t)ROWS * DIM * 2);
  __bf16* b_y  = (__bf16*)alloc((size_t)ROWS * DIM * 2);
  __bf16* b_q  = (__bf16*)alloc((size_t)ROWS * DIM * 2);
  __bf16* b_v  = (__bf16*)alloc((size_t)ROWS * DIM * 2);
  __bf16* b_cc = (__bf16*)alloc((size_t)ROWS * DIM * 2);
  __bf16* b_tmp= (__bf16*)alloc((size_t)ROWS * DIM * 2);
  __bf16* b_ff = (__bf16*)alloc((size_t)ROWS * FFD * 2);
  float*  f_x  = (float*)alloc((size_t)ROWS * DIM * 4);
  float*  f_y  = (float*)alloc((size_t)ROWS * DIM * 4);
  float*  f_t1 = (float*)alloc((size_t)ROWS * DIM * 4);

  (void)hipFuncSetAttribute((const void*)attn_decay_wmma,
                            hipFuncAttributeMaxDynamicSharedMemorySize, ATTN_SMEM);

  // ---- weights -> bf16 (every call; deterministic) ----
  cvt_f32_bf16<<<512, 256, 0, stream>>>(k_w,   Wk,  (long)NL * DIM * DIM);
  cvt_f32_bf16<<<512, 256, 0, stream>>>(v_w,   Wv,  (long)NL * DIM * DIM);
  cvt_f32_bf16<<<512, 256, 0, stream>>>(out_w, Wo,  (long)NL * DIM * DIM);
  cvt_f32_bf16<<<1024, 256, 0, stream>>>(ff1_w, Wf1, (long)NL * DIM * FFD);
  cvt_f32_bf16<<<1024, 256, 0, stream>>>(ff2_w, Wf2, (long)NL * FFD * DIM);
  cvt_f32_bf16<<<512, 256, 0, stream>>>(conv_w, Wc, (long)DIM * DIM * KSZ);

  auto proj = [&](const __bf16* A, const __bf16* B, const float* bias,
                  float* Cf, __bf16* Ch, int N, int K, int relu) {
    dim3 grd(N / 64, ROWS / 128);
    gemm_bf16_wmma<<<grd, 256, 0, stream>>>(A, B, bias, Cf, Ch, ROWS, N, K,
                                            (long)N, 1L, 0, 0, 0, relu);
  };

  // ---- smooth front-end: causal conv as 5 shifted GEMM taps ----
  auto smooth = [&](const float* emb, float* fdst, __bf16* bdst) {
    cvt_f32_bf16<<<512, 256, 0, stream>>>(emb, b_tmp, (long)ROWS * DIM);
    dim3 grd(DIM / 64, ROWS / 128);
    for (int s = 0; s < KSZ; s++) {
      // trend[o] += x[t + s-(KS-1)][i] * conv_w[o][i][s]
      gemm_bf16_wmma<<<grd, 256, 0, stream>>>(
          b_tmp, Wc + s, (s == 0) ? conv_b : nullptr, f_t1, nullptr,
          ROWS, DIM, DIM, (long)KSZ, (long)(DIM * KSZ),
          s - (KSZ - 1), SEQT, (s == 0) ? 0 : 1, 0);
    }
    smooth_ln_512<<<ROWS / 8, 256, 0, stream>>>(emb, f_t1, sqrt_b, sm_g, sm_b,
                                                fdst, bdst);
  };
  smooth(q_emb, f_x, b_x);
  smooth(qa_emb, f_y, b_y);

  // ---- transformer layer (kq_same && q_in==k_in -> single q/k projection) ----
  auto layer = [&](int i, int mtype, int ffn, float* fio, __bf16* bio,
                   const __bf16* bvsrc, float* ffinal) {
    proj(bio,   Wk + (size_t)i * DIM * DIM, k_b + i * DIM, nullptr, b_q, DIM, DIM, 0);
    proj(bvsrc, Wv + (size_t)i * DIM * DIM, v_b + i * DIM, nullptr, b_v, DIM, DIM, 0);
    attn_decay_wmma<<<dim3(SEQT / 16, NH, BSZ), 256, ATTN_SMEM, stream>>>(
        b_q, b_v, gammas, i, mtype, b_cc);
    proj(b_cc, Wo + (size_t)i * DIM * DIM, out_b + i * DIM, f_t1, nullptr, DIM, DIM, 0);
    float* ln1out = ffn ? fio : ffinal;
    add_ln_512<<<ROWS / 8, 256, 0, stream>>>(fio, f_t1, ln1_g + i * DIM,
                                             ln1_b + i * DIM, ln1out, bio, 1e-5f);
    if (ffn) {
      proj(bio,  Wf1 + (size_t)i * DIM * FFD, ff1_b + i * FFD, nullptr, b_ff, FFD, DIM, 1);
      proj(b_ff, Wf2 + (size_t)i * FFD * DIM, ff2_b + i * DIM, f_t1, nullptr, DIM, FFD, 0);
      add_ln_512<<<ROWS / 8, 256, 0, stream>>>(ln1out, f_t1, ln2_g + i * DIM,
                                               ln2_b + i * DIM, ffinal, bio, 1e-5f);
    }
  };

  // blocks_1: two self-attn+FFN layers on y
  layer(0, 1, 1, f_y, b_y, b_y, f_y);
  layer(1, 1, 1, f_y, b_y, b_y, f_y);
  // blocks_2: alternating (mask=1, no FFN) / (mask=0, values=y, FFN)
  layer(2, 1, 0, f_x, b_x, b_x, f_x);
  layer(3, 0, 1, f_x, b_x, b_y, f_x);
  layer(4, 1, 0, f_x, b_x, b_x, f_x);
  layer(5, 0, 1, f_x, b_x, b_y, (float*)d_out); // final LN writes output
}
